// AttentionHead_36112085024772
// MI455X (gfx1250) — matrix operand order repaired and checked by reading, not compiled, not run on hardware
//
#include <hip/hip_runtime.h>

typedef __attribute__((ext_vector_type(16))) _Float16 v16h;
typedef __attribute__((ext_vector_type(8)))  _Float16 v8h;
typedef __attribute__((ext_vector_type(8)))  float    v8f;
typedef __attribute__((ext_vector_type(4)))  float    v4f;

#define BB 8
#define TT 2048
#define CC 1024
#define HD 64
#define WROW 40   // padded LDS row (halves) for W^T tiles

static __device__ __forceinline__ v8f wmma_f16(v16h a, v16h b, v8f c) {
  // D = A(16x32 f16) x B(32x16 f16) + C(16x16 f32)
  return __builtin_amdgcn_wmma_f32_16x16x32_f16(false, a, false, b, (short)0, c,
                                                false, false);
}

static __device__ __forceinline__ v16h cat8(v8h lo, v8h hi) {
  v16h r;
#pragma unroll
  for (int i = 0; i < 8; i++) { r[i] = lo[i]; r[i + 8] = hi[i]; }
  return r;
}

static __device__ __forceinline__ v16h ldB16(const _Float16* p) {
  return cat8(*(const v8h*)p, *(const v8h*)(p + 16));
}

// ---------------------------------------------------------------------------
// Kernel 1: q = x@Wq, k = x@Wk, v = x@Wv (f16 outputs, v transposed).
// 8 waves/block = 4 row-tiles x 2 K-halves (split-K, merged via LDS).
// B tiles double-buffered in registers to overlap DS latency with WMMA.
// ---------------------------------------------------------------------------
__global__ __launch_bounds__(256) void proj_qkv(
    const float* __restrict__ x,  const float* __restrict__ Wq,
    const float* __restrict__ Wk, const float* __restrict__ Wv,
    _Float16* __restrict__ qo, _Float16* __restrict__ ko,
    _Float16* __restrict__ vT) {
  // arena: W^T staging (30720 B) reused for split-K merge (49152 B)
  __shared__ __align__(16) unsigned char smem[49152];
  typedef _Float16 StageT[3][HD][WROW];
  StageT* stage = (StageT*)smem;            // stage[2][3][64][WROW]
  float*  mrg   = (float*)smem;             // [4 tiles][96][32 lanes]

  const int tid    = threadIdx.x;
  const int wave   = tid >> 5;
  const int lane   = tid & 31;
  const int lanelo = lane & 15;
  const int lhalf  = lane >> 4;
  const int qt     = wave >> 1;                       // tile slot 0..3
  const int h      = wave & 1;                        // K-half
  const int tile   = blockIdx.x * 4 + qt;             // 0..1023
  const size_t arow = (size_t)tile * 16 + lanelo;     // A-layout row

  const float* Ws[3] = {Wq, Wk, Wv};
  v8f z = {};
  v8f acc[3][4];
#pragma unroll
  for (int m = 0; m < 3; m++)
#pragma unroll
    for (int t = 0; t < 4; t++) acc[m][t] = z;

  for (int i = 0; i < 16; i++) {
    // stage both K-halves' W chunks (2 x 3 x 32 x 64 f32 -> f16 transposed)
#pragma unroll
    for (int j = 0; j < 48; j++) {
      int idx = tid + j * 256;          // 0..12287
      int hh  = idx / 6144;
      int r   = idx - hh * 6144;
      int m   = r >> 11;
      int r2  = r & 2047;
      int kr  = r2 >> 6;
      int n   = r2 & 63;
      stage[hh][m][n][kr] =
          (_Float16)Ws[m][(size_t)(hh * 512 + i * 32 + kr) * HD + n];
    }
    __syncthreads();

    const int kk = h * 512 + i * 32;
    // A tile: x[tile rows][kk..kk+31] in WMMA A layout
    const float* px = x + arow * CC + kk + lhalf * 8;
    v16h a;
    {
      v4f x0 = *(const v4f*)(px);
      v4f x1 = *(const v4f*)(px + 4);
      v4f x2 = *(const v4f*)(px + 16);
      v4f x3 = *(const v4f*)(px + 20);
#pragma unroll
      for (int ii = 0; ii < 4; ii++) {
        a[ii]      = (_Float16)x0[ii];
        a[4 + ii]  = (_Float16)x1[ii];
        a[8 + ii]  = (_Float16)x2[ii];
        a[12 + ii] = (_Float16)x3[ii];
      }
    }

    // 12 (matrix, col-tile) steps, register double-buffered B
    v16h bcur = ldB16(&stage[h][0][0 * 16 + lanelo][lhalf * 8]);
#pragma unroll
    for (int s = 0; s < 12; s++) {
      const int m = s >> 2, t = s & 3;
      if (s < 11) {
        const int m2 = (s + 1) >> 2, t2 = (s + 1) & 3;
        v16h bnext = ldB16(&stage[h][m2][t2 * 16 + lanelo][lhalf * 8]);
        acc[m][t] = wmma_f16(a, bcur, acc[m][t]);
        bcur = bnext;
      } else {
        acc[m][t] = wmma_f16(a, bcur, acc[m][t]);
      }
    }
    __syncthreads();
  }

  // split-K merge through LDS (h==1 publishes, h==0 reduces + stores)
  if (h == 1) {
#pragma unroll
    for (int m = 0; m < 3; m++)
#pragma unroll
      for (int t = 0; t < 4; t++)
#pragma unroll
        for (int r = 0; r < 8; r++)
          mrg[qt * 3072 + ((m * 4 + t) * 8 + r) * 32 + lane] = acc[m][t][r];
  }
  __syncthreads();
  if (h == 0) {
#pragma unroll
    for (int m = 0; m < 3; m++)
#pragma unroll
      for (int t = 0; t < 4; t++)
#pragma unroll
        for (int r = 0; r < 8; r++)
          acc[m][t][r] += mrg[qt * 3072 + ((m * 4 + t) * 8 + r) * 32 + lane];

    const int bidx  = tile >> 7;
    const int tloc0 = (tile * 16) & (TT - 1);
#pragma unroll
    for (int t = 0; t < 4; t++) {
      const int col = t * 16 + lanelo;
#pragma unroll
      for (int r = 0; r < 8; r++) {
        size_t orow = (size_t)tile * 16 + r + 8 * lhalf;
        qo[orow * HD + col] = (_Float16)acc[0][t][r];
        ko[orow * HD + col] = (_Float16)acc[1][t][r];
      }
      v8h vp;
#pragma unroll
      for (int r = 0; r < 8; r++) vp[r] = (_Float16)acc[2][t][r];
      *(v8h*)(vT + (size_t)bidx * HD * TT + (size_t)col * TT + tloc0 +
              8 * lhalf) = vp;
    }
  }
}

// ---------------------------------------------------------------------------
// Kernel 2: flash attention. 8 waves/block = 4 query tiles x 2 key-splits;
// each wave runs online softmax over its chunk subset, pairs merge via LDS.
// ---------------------------------------------------------------------------
__global__ __launch_bounds__(256) void attn_fwd(
    const _Float16* __restrict__ qf, const _Float16* __restrict__ kf,
    const _Float16* __restrict__ vT, float* __restrict__ out) {
  __shared__ _Float16 pbuf[8][16][32];     // per-wave probability tile
  __shared__ float mrg[4][48][32];         // [tile][32 acc + 8 m + 8 l][lane]

  const int tid    = threadIdx.x;
  const int wave   = tid >> 5;
  const int lane   = tid & 31;
  const int lanelo = lane & 15;
  const int lhalf  = lane >> 4;
  const int qt     = wave >> 1;                 // tile slot 0..3
  const int h      = wave & 1;                  // key split
  const int wt     = blockIdx.x * 4 + qt;       // 0..1023
  const int qb     = wt & 127;
  const int b      = wt >> 7;
  const int qloc0  = qb * 16;

  // q block as two A tiles (K = d 0..31, 32..63)
  v16h aq[2];
  {
    const _Float16* pq = qf + ((size_t)wt * 16 + lanelo) * HD;
#pragma unroll
    for (int c = 0; c < 2; c++) {
      const _Float16* p = pq + c * 32 + lhalf * 8;
      aq[c] = cat8(*(const v8h*)p, *(const v8h*)(p + 16));
    }
  }

  v8f z = {};
  v8f acc[4];
  float mi[8], li[8];
#pragma unroll
  for (int t = 0; t < 4; t++) acc[t] = z;
#pragma unroll
  for (int r = 0; r < 8; r++) { mi[r] = -__builtin_inff(); li[r] = 0.f; }

  const int nchunk = (qloc0 + 15) / 32 + 1;
  const _Float16* kbb = kf + (size_t)b * TT * HD;
  const _Float16* vbb = vT + (size_t)b * HD * TT;

  for (int c = h; c < nchunk; c += 2) {
    const int jbase = c * 32;

    // issue ALL global B loads for this chunk up front (k tiles + v tiles)
    v16h bk[2][2];
#pragma unroll
    for (int ct = 0; ct < 2; ct++) {
      const _Float16* pk = kbb + (size_t)(jbase + ct * 16 + lanelo) * HD;
#pragma unroll
      for (int kc = 0; kc < 2; kc++)
        bk[ct][kc] = ldB16(pk + kc * 32 + lhalf * 8);
    }
    v16h bv[4];
#pragma unroll
    for (int dt = 0; dt < 4; dt++)
      bv[dt] = ldB16(vbb + (size_t)(dt * 16 + lanelo) * TT + jbase + lhalf * 8);

    // scores
    v8f s[2];
#pragma unroll
    for (int ct = 0; ct < 2; ct++)
      s[ct] = wmma_f16(aq[1], bk[ct][1], wmma_f16(aq[0], bk[ct][0], z));

    // causal mask (-inf) then scale by 1/sqrt(C)=1/32 (reference order)
    float pm[8], e0[8], e1[8];
    const int j0 = jbase + lanelo;
    const int j1 = j0 + 16;
#pragma unroll
    for (int r = 0; r < 8; r++) {
      const int qi = qloc0 + r + 8 * lhalf;
      float a  = (j0 <= qi) ? s[0][r] * 0.03125f : -__builtin_inff();
      float bb = (j1 <= qi) ? s[1][r] * 0.03125f : -__builtin_inff();
      s[0][r] = a; s[1][r] = bb;
      pm[r] = fmaxf(a, bb);
    }
#pragma unroll
    for (int off = 1; off < 16; off <<= 1)
#pragma unroll
      for (int r = 0; r < 8; r++)
        pm[r] = fmaxf(pm[r], __shfl_xor(pm[r], off, 32));

#pragma unroll
    for (int r = 0; r < 8; r++) {
      float mn    = fmaxf(mi[r], pm[r]);
      float alpha = __expf(mi[r] - mn);
      mi[r] = mn;
      e0[r] = __expf(s[0][r] - mn);
      e1[r] = __expf(s[1][r] - mn);
      pm[r] = e0[r] + e1[r];
      li[r] *= alpha;
#pragma unroll
      for (int t = 0; t < 4; t++) acc[t][r] *= alpha;
    }
#pragma unroll
    for (int off = 1; off < 16; off <<= 1)
#pragma unroll
      for (int r = 0; r < 8; r++) pm[r] += __shfl_xor(pm[r], off, 32);
#pragma unroll
    for (int r = 0; r < 8; r++) li[r] += pm[r];

    // relayout p: C/D layout -> LDS row-major -> A layout (wave-private)
#pragma unroll
    for (int r = 0; r < 8; r++) {
      const int gr = r + 8 * lhalf;
      pbuf[wave][gr][lanelo]      = (_Float16)e0[r];
      pbuf[wave][gr][16 + lanelo] = (_Float16)e1[r];
    }
    asm volatile("s_wait_dscnt 0" ::: "memory");
    v16h ap;
    {
      const _Float16* pp = &pbuf[wave][lanelo][lhalf * 8];
      ap = cat8(*(const v8h*)pp, *(const v8h*)(pp + 16));
    }

    // acc += p @ v
#pragma unroll
    for (int dt = 0; dt < 4; dt++)
      acc[dt] = wmma_f16(ap, bv[dt], acc[dt]);
  }

  // two-way flash merge through LDS: h==1 publishes (m,l,acc); h==0 combines
  if (h == 1) {
#pragma unroll
    for (int t = 0; t < 4; t++)
#pragma unroll
      for (int r = 0; r < 8; r++) mrg[qt][t * 8 + r][lane] = acc[t][r];
#pragma unroll
    for (int r = 0; r < 8; r++) {
      mrg[qt][32 + r][lane] = mi[r];
      mrg[qt][40 + r][lane] = li[r];
    }
  }
  __syncthreads();
  if (h == 0) {
#pragma unroll
    for (int r = 0; r < 8; r++) {
      const float mB = mrg[qt][32 + r][lane];
      const float lB = mrg[qt][40 + r][lane];
      const float mn = fmaxf(mi[r], mB);
      const float a0 = __expf(mi[r] - mn);
      const float a1 = __expf(mB - mn);
      const float l  = li[r] * a0 + lB * a1;
      const float inv = 1.0f / l;
      size_t orow = (size_t)wt * 16 + r + 8 * lhalf;
#pragma unroll
      for (int t = 0; t < 4; t++)
        out[orow * HD + t * 16 + lanelo] =
            (acc[t][r] * a0 + mrg[qt][t * 8 + r][lane] * a1) * inv;
    }
  }
}

// ---------------------------------------------------------------------------
extern "C" void kernel_launch(void* const* d_in, const int* in_sizes, int n_in,
                              void* d_out, int out_size, void* d_ws,
                              size_t ws_size, hipStream_t stream) {
  (void)in_sizes; (void)n_in; (void)out_size; (void)ws_size;
  const float* x  = (const float*)d_in[0];
  const float* Wq = (const float*)d_in[1];
  const float* Wk = (const float*)d_in[2];
  const float* Wv = (const float*)d_in[3];

  const size_t N = (size_t)BB * TT * HD;     // 1,048,576 halves each
  _Float16* qf = (_Float16*)d_ws;            // 2 MiB
  _Float16* kf = qf + N;                     // 2 MiB
  _Float16* vT = kf + N;                     // 2 MiB (transposed [b][d][t])

  const int blocks = (BB * TT / 16) / 4;     // 256 blocks x 8 waves
  proj_qkv<<<blocks, 256, 0, stream>>>(x, Wq, Wk, Wv, qf, kf, vT);
  attn_fwd<<<blocks, 256, 0, stream>>>(qf, kf, vT, (float*)d_out);
}
